// ParallelProsodyPredictor_27968827031684
// MI455X (gfx1250) — matrix operand order repaired and checked by reading, not compiled, not run on hardware
//
#include <hip/hip_runtime.h>
#include <cmath>

// ---------------- problem constants ----------------
constexpr int NB  = 64;      // batch
constexpr int TT  = 1024;    // time steps
constexpr int EE  = 256;     // embedding
constexpr int HH  = 128;     // per-direction hidden
constexpr int GG  = 3 * HH;  // stacked gates [r,z,n] = 384
constexpr int BOT = 32;      // bottleneck
constexpr long NT = (long)NB * TT;  // 65536 rows

typedef __attribute__((ext_vector_type(16))) __bf16 v16bf;
typedef __attribute__((ext_vector_type(8)))  __bf16 v8bf;
typedef __attribute__((ext_vector_type(4)))  __bf16 v4bf;
typedef __attribute__((ext_vector_type(8)))  float  v8f;

// ---------------------------------------------------------------------------
// Kernel 1: gx GEMM.  C[65536 x 768] = X[65536 x 256] @ W^T, W = [w_ih_f; w_ih_b]
// 128x128 tile per workgroup, 8 waves in 2x4.  fwd/bwd select hoisted to
// scalars (uniform per block).  Epilogue transposes through LDS for fully
// coalesced b128 global stores.
// ---------------------------------------------------------------------------
__global__ __launch_bounds__(256)
void gx_gemm_kernel(const float* __restrict__ x,
                    const float* __restrict__ w_f, const float* __restrict__ w_b,
                    const float* __restrict__ bi_f, const float* __restrict__ bi_b,
                    __bf16* __restrict__ gx_f, __bf16* __restrict__ gx_b)
{
    __shared__ __align__(16) __bf16 lx[128][36];   // [row][k], 72B row stride
    __shared__ __align__(16) __bf16 lw[128][36];   // [col][k]
    __shared__ __align__(16) __bf16 lc[128][136];  // epilogue tile, 272B stride

    const int tid  = threadIdx.x;
    const int lane = tid & 31;
    const int wid  = tid >> 5;
    const int rowBase = blockIdx.x * 128;   // over NT
    const int colBase = blockIdx.y * 128;   // over 2*GG = 768
    const int wy = wid >> 2;                // 0..1
    const int wx = wid & 3;                 // 0..3

    // uniform fwd/bwd select (colBase is a multiple of 128; GG = 384)
    const bool isB = (colBase >= GG);
    const float*  wsel  = isB ? w_b  : w_f;
    const float*  bsel  = isB ? bi_b : bi_f;
    __bf16*       gxsel = isB ? gx_b : gx_f;
    const int     colAdj = colBase - (isB ? GG : 0);

    v8f acc[4][2];
#pragma unroll
    for (int i = 0; i < 4; ++i)
#pragma unroll
        for (int j = 0; j < 2; ++j) acc[i][j] = (v8f)0.0f;

    const int arowsel = lane & 15;
    const int akoff   = (lane >> 4) << 4;   // 0 or 16
    const int bcolsel = lane & 15;
    const int bkoff   = (lane >> 4) << 4;   // 0 or 16

    for (int k0 = 0; k0 < EE; k0 += 32) {
        // ---- stage X and W tiles: 1024 float4 slots each, 4 per thread
        float4 tx[4], tw[4];
#pragma unroll
        for (int i = 0; i < 4; ++i) {
            int s  = tid + i * 256;
            int r  = s >> 3;            // 8 float4 per row
            int kq = (s & 7) << 2;
            tx[i] = *(const float4*)&x[(size_t)(rowBase + r) * EE + k0 + kq];
        }
#pragma unroll
        for (int i = 0; i < 4; ++i) {
            int s  = tid + i * 256;
            int c  = s >> 3;
            int kq = (s & 7) << 2;
            tw[i] = *(const float4*)&wsel[(size_t)(colAdj + c) * EE + k0 + kq];
        }
        // prefetch next K-slab of x, keep near (global_prefetch_b8)
        if (k0 + 32 < EE)
            __builtin_prefetch(&x[(size_t)(rowBase + (tid >> 1)) * EE + k0 + 32 + (tid & 1) * 16], 0, 3);
        // pack fp32 -> bf16, 8-byte LDS stores
#pragma unroll
        for (int i = 0; i < 4; ++i) {
            int s  = tid + i * 256;
            int r  = s >> 3;
            int kq = (s & 7) << 2;
            v4bf p;
            p[0] = (__bf16)tx[i].x; p[1] = (__bf16)tx[i].y;
            p[2] = (__bf16)tx[i].z; p[3] = (__bf16)tx[i].w;
            *(v4bf*)&lx[r][kq] = p;
            v4bf q;
            q[0] = (__bf16)tw[i].x; q[1] = (__bf16)tw[i].y;
            q[2] = (__bf16)tw[i].z; q[3] = (__bf16)tw[i].w;
            *(v4bf*)&lw[r][kq] = q;
        }
        __syncthreads();

        // A fragments (one per row tile)
        v16bf a[4];
#pragma unroll
        for (int i = 0; i < 4; ++i) {
            int r = wy * 64 + i * 16 + arowsel;
#pragma unroll
            for (int q = 0; q < 8; ++q) {
                a[i][q]     = lx[r][akoff + q];
                a[i][q + 8] = lx[r][akoff + 8 + q];
            }
        }
#pragma unroll
        for (int j = 0; j < 2; ++j) {
            int c = wx * 32 + j * 16 + bcolsel;
            v16bf b;
#pragma unroll
            for (int q = 0; q < 16; ++q) b[q] = lw[c][bkoff + q];
#pragma unroll
            for (int i = 0; i < 4; ++i)
                acc[i][j] = __builtin_amdgcn_wmma_f32_16x16x32_bf16(
                    false, a[i], false, b, (short)0, acc[i][j], false, false);
        }
        __syncthreads();
    }

    // ---- epilogue: +bias into LDS tile, then coalesced b128 global stores
    const int crow = (lane >> 4) << 3;  // M = v + crow
    const int ccol = lane & 15;
    float bj[2];
#pragma unroll
    for (int j = 0; j < 2; ++j) bj[j] = bsel[colAdj + wx * 32 + j * 16 + ccol];
#pragma unroll
    for (int i = 0; i < 4; ++i)
#pragma unroll
        for (int j = 0; j < 2; ++j)
#pragma unroll
            for (int v = 0; v < 8; ++v)
                lc[wy * 64 + i * 16 + crow + v][wx * 32 + j * 16 + ccol] =
                    (__bf16)(acc[i][j][v] + bj[j]);
    __syncthreads();
    // 2048 v8bf slots, 8 per thread
#pragma unroll
    for (int i = 0; i < 8; ++i) {
        int e  = tid + i * 256;
        int r  = e >> 4;            // 16 v8bf per row
        int c8 = (e & 15) << 3;
        *(v8bf*)&gxsel[(size_t)(rowBase + r) * GG + colAdj + c8] = *(const v8bf*)&lc[r][c8];
    }
}

// ---------------------------------------------------------------------------
// Kernel 2: sequential GRU recurrence.  8 blocks = 2 dirs x 4 batch-tiles(16).
// W_hh B-fragments register-resident.  h_prev lives in thread registers
// (producer == consumer); LDS holds h only in bf16 fragment layout so each
// A fragment is two ds_load_b128 per k-tile.
// ---------------------------------------------------------------------------
__global__ __launch_bounds__(256)
void gru_recurrence_kernel(const __bf16* __restrict__ gx_f, const __bf16* __restrict__ gx_b,
                           const float* __restrict__ whh_f, const float* __restrict__ whh_b,
                           const float* __restrict__ bhh_f, const float* __restrict__ bhh_b,
                           __bf16* __restrict__ h_f, __bf16* __restrict__ h_b)
{
    const int dir = blockIdx.x >> 2;        // 0 = fwd, 1 = bwd
    const int n0  = (blockIdx.x & 3) * 16;  // batch-tile base row

    const __bf16* gx  = dir ? gx_b  : gx_f;
    const float*  whh = dir ? whh_b : whh_f;
    const float*  bhh = dir ? bhh_b : bhh_f;
    __bf16*       hout= dir ? h_b   : h_f;

    __shared__ __align__(16) __bf16 lhb[16][136];  // bf16 hidden, 272B row stride
    __shared__ float lgh[16][GG];                  // gate pre-activations

    const int tid  = threadIdx.x;
    const int lane = tid & 31;
    const int wid  = tid >> 5;       // 0..7, owns gate cols [wid*48, wid*48+48)

    // ---- preload step-invariant W_hh B-fragments into registers ----
    const int bcol0 = wid * 48 + (lane & 15);
    const int bk0   = (lane >> 4) << 4;
    v16bf bw[3][4];
#pragma unroll
    for (int j = 0; j < 3; ++j)
#pragma unroll
        for (int kt = 0; kt < 4; ++kt) {
            const float* wp = &whh[(size_t)(bcol0 + j * 16) * HH + kt * 32 + bk0];
            float4 t0 = *(const float4*)&wp[0];
            float4 t1 = *(const float4*)&wp[4];
            float4 t2 = *(const float4*)&wp[8];
            float4 t3 = *(const float4*)&wp[12];
            bw[j][kt][0]  = (__bf16)t0.x; bw[j][kt][1]  = (__bf16)t0.y;
            bw[j][kt][2]  = (__bf16)t0.z; bw[j][kt][3]  = (__bf16)t0.w;
            bw[j][kt][4]  = (__bf16)t1.x; bw[j][kt][5]  = (__bf16)t1.y;
            bw[j][kt][6]  = (__bf16)t1.z; bw[j][kt][7]  = (__bf16)t1.w;
            bw[j][kt][8]  = (__bf16)t2.x; bw[j][kt][9]  = (__bf16)t2.y;
            bw[j][kt][10] = (__bf16)t2.z; bw[j][kt][11] = (__bf16)t2.w;
            bw[j][kt][12] = (__bf16)t3.x; bw[j][kt][13] = (__bf16)t3.y;
            bw[j][kt][14] = (__bf16)t3.z; bw[j][kt][15] = (__bf16)t3.w;
        }

    const int ccol = lane & 15;
    const int crow = (lane >> 4) << 3;
    float bias[3];
#pragma unroll
    for (int j = 0; j < 3; ++j) bias[j] = bhh[wid * 48 + j * 16 + ccol];

    // gate-phase mapping: each thread owns row er, 8 contiguous columns at ec
    const int er = tid >> 4;          // row 0..15
    const int ec = (tid & 15) << 3;   // col base, multiple of 8

    // h0 = 0 (registers + LDS bf16 copy)
    float hreg[8];
#pragma unroll
    for (int q = 0; q < 8; ++q) hreg[q] = 0.0f;
    *(v8bf*)&lhb[er][ec] = (v8bf)(__bf16)0.0f;
    __syncthreads();

    const int arow  = lane & 15;
    const int akoff = (lane >> 4) << 4;

    for (int s = 0; s < TT; ++s) {
        const int t = dir ? (TT - 1 - s) : s;

        // issue gx loads for this step early (hide under WMMA)
        const size_t gbase = ((size_t)(n0 + er) * TT + t) * GG;
        v8bf gvr = *(const v8bf*)&gx[gbase + ec];
        v8bf gvz = *(const v8bf*)&gx[gbase + 128 + ec];
        v8bf gvn = *(const v8bf*)&gx[gbase + 256 + ec];

        // A fragments: two b128 LDS loads per k-tile
        v16bf a[4];
#pragma unroll
        for (int kt = 0; kt < 4; ++kt) {
            v8bf lo = *(const v8bf*)&lhb[arow][kt * 32 + akoff];
            v8bf hi = *(const v8bf*)&lhb[arow][kt * 32 + akoff + 8];
#pragma unroll
            for (int q = 0; q < 8; ++q) { a[kt][q] = lo[q]; a[kt][q + 8] = hi[q]; }
        }

        v8f acc[3];
#pragma unroll
        for (int j = 0; j < 3; ++j) acc[j] = (v8f)0.0f;
#pragma unroll
        for (int kt = 0; kt < 4; ++kt)
#pragma unroll
            for (int j = 0; j < 3; ++j)
                acc[j] = __builtin_amdgcn_wmma_f32_16x16x32_bf16(
                    false, a[kt], false, bw[j][kt], (short)0, acc[j], false, false);

        // gh (+ b_hh) -> LDS
#pragma unroll
        for (int j = 0; j < 3; ++j)
#pragma unroll
            for (int v = 0; v < 8; ++v)
                lgh[crow + v][wid * 48 + j * 16 + ccol] = acc[j][v] + bias[j];
        __syncthreads();

        // prefetch next step's gx while gate math runs
        if (s + 1 < TT) {
            const int tn = dir ? (t - 1) : (t + 1);
            const size_t gn = ((size_t)(n0 + er) * TT + tn) * GG;
            __builtin_prefetch(&gx[gn + ec], 0, 3);
            __builtin_prefetch(&gx[gn + 128 + ec], 0, 3);
            __builtin_prefetch(&gx[gn + 256 + ec], 0, 3);
        }

        // gate math: h_prev in registers (f32)
        v8bf hv;
#pragma unroll
        for (int q = 0; q < 8; ++q) {
            int c = ec + q;
            float gxr = (float)gvr[q];
            float gxz = (float)gvz[q];
            float gxn = (float)gvn[q];
            float ghr = lgh[er][c];
            float ghz = lgh[er][128 + c];
            float ghn = lgh[er][256 + c];
            float rg  = 1.0f / (1.0f + expf(-(gxr + ghr)));
            float zg  = 1.0f / (1.0f + expf(-(gxz + ghz)));
            float ng  = tanhf(gxn + rg * ghn);
            float hn  = (1.0f - zg) * ng + zg * hreg[q];
            hreg[q]   = hn;
            hv[q]     = (__bf16)hn;
        }
        *(v8bf*)&lhb[er][ec] = hv;                                     // for next A-frags
        *(v8bf*)&hout[((size_t)(n0 + er) * TT + t) * HH + ec] = hv;    // sequence output
        __syncthreads();
    }
}

// ---------------------------------------------------------------------------
// Kernel 3: out[65536 x 32] = concat(h_f, h_b)[65536 x 256] @ w_out^T + b_out
// 8 waves x 4 row-tiles x 2 col-tiles per workgroup (512 rows per block).
// ---------------------------------------------------------------------------
__global__ __launch_bounds__(256)
void out_proj_kernel(const __bf16* __restrict__ h_f, const __bf16* __restrict__ h_b,
                     const float* __restrict__ w_out, const float* __restrict__ b_out,
                     float* __restrict__ out)
{
    const int tid  = threadIdx.x;
    const int lane = tid & 31;
    const int wid  = tid >> 5;
    const int rowBase = blockIdx.x * 512 + wid * 64;

    const int arow  = lane & 15;
    const int akoff = (lane >> 4) << 4;
    const int bcol  = lane & 15;
    const int bkoff = (lane >> 4) << 4;

    v8f acc[4][2];
#pragma unroll
    for (int i = 0; i < 4; ++i)
#pragma unroll
        for (int j = 0; j < 2; ++j) acc[i][j] = (v8f)0.0f;

#pragma unroll
    for (int kt = 0; kt < 8; ++kt) {
        const int kg0 = kt * 32;
        // B fragments: B[k][j] = w_out[j][k]  (w_out is [32][256] row-major)
        v16bf b[2];
#pragma unroll
        for (int j = 0; j < 2; ++j) {
            const float* wp = &w_out[(size_t)(j * 16 + bcol) * EE + kg0 + bkoff];
            float4 t0 = *(const float4*)&wp[0];
            float4 t1 = *(const float4*)&wp[4];
            float4 t2 = *(const float4*)&wp[8];
            float4 t3 = *(const float4*)&wp[12];
            b[j][0]  = (__bf16)t0.x; b[j][1]  = (__bf16)t0.y;
            b[j][2]  = (__bf16)t0.z; b[j][3]  = (__bf16)t0.w;
            b[j][4]  = (__bf16)t1.x; b[j][5]  = (__bf16)t1.y;
            b[j][6]  = (__bf16)t1.z; b[j][7]  = (__bf16)t1.w;
            b[j][8]  = (__bf16)t2.x; b[j][9]  = (__bf16)t2.y;
            b[j][10] = (__bf16)t2.z; b[j][11] = (__bf16)t2.w;
            b[j][12] = (__bf16)t3.x; b[j][13] = (__bf16)t3.y;
            b[j][14] = (__bf16)t3.z; b[j][15] = (__bf16)t3.w;
        }

        const __bf16* hsrc = (kg0 < HH) ? h_f : h_b;
        const int koff     = (kg0 < HH) ? kg0 : (kg0 - HH);
#pragma unroll
        for (int i = 0; i < 4; ++i) {
            size_t rbase = (size_t)(rowBase + i * 16 + arow) * HH;
            v8bf lo = *(const v8bf*)&hsrc[rbase + koff + akoff];
            v8bf hi = *(const v8bf*)&hsrc[rbase + koff + akoff + 8];
            v16bf a;
#pragma unroll
            for (int q = 0; q < 8; ++q) { a[q] = lo[q]; a[q + 8] = hi[q]; }
#pragma unroll
            for (int j = 0; j < 2; ++j)
                acc[i][j] = __builtin_amdgcn_wmma_f32_16x16x32_bf16(
                    false, a, false, b[j], (short)0, acc[i][j], false, false);
        }
    }

    const int crow = (lane >> 4) << 3;
    const int ccol = lane & 15;
#pragma unroll
    for (int i = 0; i < 4; ++i)
#pragma unroll
        for (int j = 0; j < 2; ++j)
#pragma unroll
            for (int v = 0; v < 8; ++v)
                out[(size_t)(rowBase + i * 16 + crow + v) * BOT + j * 16 + ccol] =
                    acc[i][j][v] + b_out[j * 16 + ccol];
}

// ---------------------------------------------------------------------------
extern "C" void kernel_launch(void* const* d_in, const int* in_sizes, int n_in,
                              void* d_out, int out_size, void* d_ws, size_t ws_size,
                              hipStream_t stream) {
    (void)in_sizes; (void)n_in; (void)out_size; (void)ws_size;
    const float* x      = (const float*)d_in[0];
    const float* w_ih_f = (const float*)d_in[1];
    const float* w_hh_f = (const float*)d_in[2];
    const float* b_ih_f = (const float*)d_in[3];
    const float* b_hh_f = (const float*)d_in[4];
    const float* w_ih_b = (const float*)d_in[5];
    const float* w_hh_b = (const float*)d_in[6];
    const float* b_ih_b = (const float*)d_in[7];
    const float* b_hh_b = (const float*)d_in[8];
    const float* w_out  = (const float*)d_in[9];
    const float* b_out  = (const float*)d_in[10];
    float* out = (float*)d_out;

    // workspace layout (bf16): gx_f[NT*384] | gx_b[NT*384] | h_f[NT*128] | h_b[NT*128]
    __bf16* gx_f = (__bf16*)d_ws;
    __bf16* gx_b = gx_f + (size_t)NT * GG;
    __bf16* h_f  = gx_b + (size_t)NT * GG;
    __bf16* h_b  = h_f + (size_t)NT * HH;

    dim3 g1(NT / 128, (2 * GG) / 128);  // 512 x 6
    gx_gemm_kernel<<<g1, 256, 0, stream>>>(x, w_ih_f, w_ih_b, b_ih_f, b_ih_b, gx_f, gx_b);
    gru_recurrence_kernel<<<8, 256, 0, stream>>>(gx_f, gx_b, w_hh_f, w_hh_b,
                                                 b_hh_f, b_hh_b, h_f, h_b);
    out_proj_kernel<<<NT / 512, 256, 0, stream>>>(h_f, h_b, w_out, b_out, out);
}
